// FusedHadamardLinear_14190571946378
// MI455X (gfx1250) — matrix operand order repaired and checked by reading, not compiled, not run on hardware
//
#include <hip/hip_runtime.h>

typedef __attribute__((ext_vector_type(16))) _Float16 v16h;
typedef __attribute__((ext_vector_type(8)))  _Float16 v8h;
typedef __attribute__((ext_vector_type(8)))  float    v8f;
typedef __attribute__((ext_vector_type(4)))  unsigned int v4u;
typedef __attribute__((ext_vector_type(8)))  int      v8i;
typedef __attribute__((ext_vector_type(4)))  int      v4i;

#define N_FEAT 4096
#define TOKENS 8192
#define BM 128
#define BN 256
#define BK 32
#define LDS_STRIDE 40   // BK + 8 halves padding (80B row stride; TDM pad reproduces this)

#if defined(__gfx1250__) && __has_builtin(__builtin_amdgcn_tensor_load_to_lds)
#define USE_TDM 1
#else
#define USE_TDM 0
#endif

// ---------------------------------------------------------------------------
// Row-wise FWHT (length 4096): 8 stages in LDS + 4 stages in registers.
// One block = one row, 256 threads.
// ---------------------------------------------------------------------------
__device__ inline void fwht_lds_stages(float* row, int t)
{
  for (int h = 1; h <= 128; h <<= 1) {
#pragma unroll
    for (int p = 0; p < 8; ++p) {
      const int pr = t + p * 256;                       // pair index 0..2047
      const int i  = ((pr & ~(h - 1)) << 1) | (pr & (h - 1));
      const float a = row[i], b = row[i + h];
      row[i]     = a + b;
      row[i + h] = a - b;
    }
    __syncthreads();
  }
}

__device__ inline void fwht_reg_stages(float* r)
{
#pragma unroll
  for (int s = 1; s < 16; s <<= 1) {                    // h = 256*s
#pragma unroll
    for (int i = 0; i < 16; ++i) {
      if ((i & s) == 0) {
        const float a = r[i], b = r[i + s];
        r[i]     = a + b;
        r[i + s] = a - b;
      }
    }
  }
}

__global__ __launch_bounds__(256) void fwht_rows_f32_to_f16(
    const float* __restrict__ in, _Float16* __restrict__ out)
{
  __shared__ float row[N_FEAT];
  const int t = threadIdx.x;
  const float4* src = (const float4*)(in + (size_t)blockIdx.x * N_FEAT);
  float4* rowv = (float4*)row;
#pragma unroll
  for (int i = 0; i < 4; ++i) rowv[t + i * 256] = src[t + i * 256];
  __syncthreads();

  fwht_lds_stages(row, t);

  float r[16];
#pragma unroll
  for (int i = 0; i < 16; ++i) r[i] = row[t + 256 * i];
  fwht_reg_stages(r);

  _Float16* dst = out + (size_t)blockIdx.x * N_FEAT;
#pragma unroll
  for (int i = 0; i < 16; ++i)
    dst[t + 256 * i] = (_Float16)(r[i] * 0.015625f);    // 1/sqrt(4096)
}

__global__ __launch_bounds__(256) void fwht_rows_f32_inplace(float* __restrict__ data)
{
  __shared__ float row[N_FEAT];
  const int t = threadIdx.x;
  float4* g = (float4*)(data + (size_t)blockIdx.x * N_FEAT);
  float4* rowv = (float4*)row;
#pragma unroll
  for (int i = 0; i < 4; ++i) rowv[t + i * 256] = g[t + i * 256];
  __syncthreads();

  fwht_lds_stages(row, t);

  float r[16];
#pragma unroll
  for (int i = 0; i < 16; ++i) r[i] = row[t + 256 * i];
  fwht_reg_stages(r);

  float* d = data + (size_t)blockIdx.x * N_FEAT;
#pragma unroll
  for (int i = 0; i < 16; ++i)
    d[t + 256 * i] = r[i] * 0.015625f;
}

// ---------------------------------------------------------------------------
// f32 -> f16 elementwise convert (weight). 8 elements / thread.
// ---------------------------------------------------------------------------
__global__ __launch_bounds__(256) void cvt_f32_f16(
    const float* __restrict__ in, _Float16* __restrict__ out)
{
  const size_t base = ((size_t)blockIdx.x * 256 + threadIdx.x) * 8;
  const float4 a = ((const float4*)(in + base))[0];
  const float4 b = ((const float4*)(in + base))[1];
  v8h h;
  h[0] = (_Float16)a.x; h[1] = (_Float16)a.y;
  h[2] = (_Float16)a.z; h[3] = (_Float16)a.w;
  h[4] = (_Float16)b.x; h[5] = (_Float16)b.y;
  h[6] = (_Float16)b.z; h[7] = (_Float16)b.w;
  *(v8h*)(out + base) = h;
}

// ---------------------------------------------------------------------------
// TDM: 2D tile load, `rows` x 64B, global row stride 8KB (N_FEAT halves),
// LDS padded to 80B rows (pad_interval: 16 DW before pad, pad_amount: 4 DW).
// Descriptor layout per CDNA5 ISA 8.3/8.4 (D# group 0 + group 1).
// ---------------------------------------------------------------------------
#if USE_TDM
__device__ inline void tdm_load_tile(unsigned lds_addr, unsigned long long gaddr,
                                     unsigned rows, unsigned tensor_rows)
{
  const unsigned tensor_dim0 = (N_FEAT * 2u) / 8u;     // row length in 8B units
  v4u g0;
  g0[0] = 1u;                                          // count=1, user mode
  g0[1] = lds_addr;                                    // LDS byte address
  g0[2] = (unsigned)(gaddr & 0xFFFFFFFFull);           // global_addr[31:0]
  g0[3] = ((unsigned)(gaddr >> 32) & 0x01FFFFFFu)      // global_addr[56:32]
          | 0x80000000u;                               // type=2 ("image")
  v8i g1;
  // data_size [17:16]=3 (8B), pad_enable [20], pad_interval [24:22]=3 (16 DW),
  // pad_amount [31:25]=3 (4 DW)
  g1[0] = (int)((3u << 16) | (1u << 20) | (3u << 22) | (3u << 25));
  g1[1] = (int)((tensor_dim0 & 0xFFFFu) << 16);        // tensor_dim0[15:0]
  g1[2] = (int)(((tensor_dim0 >> 16) & 0xFFFFu) |
                ((tensor_rows & 0xFFFFu) << 16));      // dim0 hi | tensor_dim1 lo
  g1[3] = (int)(((tensor_rows >> 16) & 0xFFFFu) |
                (8u << 16));                           // dim1 hi | tile_dim0=8 (64B)
  g1[4] = (int)(rows & 0xFFFFu);                       // tile_dim1 = rows, tile_dim2=0
  g1[5] = (int)tensor_dim0;                            // tensor_dim0_stride lo (=row len)
  g1[6] = 0;                                           // stride hi | dim1_stride lo
  g1[7] = 0;
  const v4i gz4 = {0, 0, 0, 0};                        // 2D tensor: groups 2/3 unused
  const v8i gz8 = {0, 0, 0, 0, 0, 0, 0, 0};
  __builtin_amdgcn_tensor_load_to_lds(g0, g1, gz4, gz4, gz8, 0);
}

__device__ inline unsigned lds_byte_off(const void* p)
{
  // flat LDS aperture: addr[31:0] is the LDS byte offset
  return (unsigned)(unsigned long long)(uintptr_t)p;
}
#endif

// ---------------------------------------------------------------------------
// NT GEMM: C[t,o] = sum_k A[t,k]*B[o,k] + bias[o]
// A: [TOKENS,N_FEAT] f16, B: [N_FEAT,N_FEAT] f16, C f32.
// Block 256 thr (8 waves 2x4), block tile 128x256, BK=32, double-buffered LDS.
// Wave tile 64x64 -> 16 WMMA accumulators.
// ---------------------------------------------------------------------------
__global__ __launch_bounds__(256)
void gemm_wmma_f16(const _Float16* __restrict__ A,
                   const _Float16* __restrict__ B,
                   const float* __restrict__ bias,
                   float* __restrict__ C)
{
  __shared__ __align__(16) _Float16 sA[2][BM * LDS_STRIDE];   // 20 KB
  __shared__ __align__(16) _Float16 sB[2][BN * LDS_STRIDE];   // 40 KB

  const int tid  = threadIdx.x;
  const int lane = tid & 31;
  const int wid  = tid >> 5;
  const int wm   = wid >> 2;          // 0..1  (64 M-rows each)
  const int wn   = wid & 3;           // 0..3  (64 N-cols each)
  const size_t rowBase = (size_t)blockIdx.y * BM;
  const size_t colBase = (size_t)blockIdx.x * BN;

  // WMMA fragment addressing (16-bit A 16x32 / B 32x16 layouts)
  const int frow  = lane & 15;
  const int kselA = (lane >> 4) * 8;  // A: 8-half chunks at {0,16} / {8,24}
  const int kselB = (lane >> 4) * 8;  // B: 8-half chunks at {0,16} / {8,24}

  v8f acc[4][4] = {};

  const _Float16* aTile = A + rowBase * N_FEAT;
  const _Float16* bTile = B + colBase * N_FEAT;

#if USE_TDM
  if (wid == 0) {
    tdm_load_tile(lds_byte_off(&sA[0][0]),
                  (unsigned long long)(uintptr_t)aTile, BM, TOKENS);
    tdm_load_tile(lds_byte_off(&sB[0][0]),
                  (unsigned long long)(uintptr_t)bTile, BN, N_FEAT);
  }
#else
  // synchronous staging indices: A -> 1 segment/thread, B -> 2 segments/thread
  const int aRow = tid >> 1;
  const int aCol = (tid & 1) * 16;
#endif

  int buf = 0;
  for (int k0 = 0; k0 < N_FEAT; k0 += BK) {
#if USE_TDM
    if (wid == 0) __builtin_amdgcn_s_wait_tensorcnt(0);
    __syncthreads();                          // tile `buf` visible to all waves
    if (k0 + BK < N_FEAT && wid == 0) {       // kick next tile; overlaps compute
      tdm_load_tile(lds_byte_off(&sA[buf ^ 1][0]),
                    (unsigned long long)(uintptr_t)(aTile + k0 + BK), BM, TOKENS);
      tdm_load_tile(lds_byte_off(&sB[buf ^ 1][0]),
                    (unsigned long long)(uintptr_t)(bTile + k0 + BK), BN, N_FEAT);
    }
#else
    {
      const v8h ga0 = *(const v8h*)(aTile + (size_t)aRow * N_FEAT + k0 + aCol);
      const v8h ga1 = *(const v8h*)(aTile + (size_t)aRow * N_FEAT + k0 + aCol + 8);
      const v8h gb0 = *(const v8h*)(bTile + (size_t)aRow * N_FEAT + k0 + aCol);
      const v8h gb1 = *(const v8h*)(bTile + (size_t)aRow * N_FEAT + k0 + aCol + 8);
      const v8h gb2 = *(const v8h*)(bTile + (size_t)(aRow + 128) * N_FEAT + k0 + aCol);
      const v8h gb3 = *(const v8h*)(bTile + (size_t)(aRow + 128) * N_FEAT + k0 + aCol + 8);
      *(v8h*)&sA[buf][aRow * LDS_STRIDE + aCol]             = ga0;
      *(v8h*)&sA[buf][aRow * LDS_STRIDE + aCol + 8]         = ga1;
      *(v8h*)&sB[buf][aRow * LDS_STRIDE + aCol]             = gb0;
      *(v8h*)&sB[buf][aRow * LDS_STRIDE + aCol + 8]         = gb1;
      *(v8h*)&sB[buf][(aRow + 128) * LDS_STRIDE + aCol]     = gb2;
      *(v8h*)&sB[buf][(aRow + 128) * LDS_STRIDE + aCol + 8] = gb3;
      if (k0 + BK < N_FEAT) {
        __builtin_prefetch(aTile + (size_t)aRow * N_FEAT + k0 + BK + aCol, 0, 3);
        __builtin_prefetch(bTile + (size_t)aRow * N_FEAT + k0 + BK + aCol, 0, 3);
      }
    }
    __syncthreads();
#endif

    // fragments
    v16h afrag[4], bfrag[4];
#pragma unroll
    for (int mt = 0; mt < 4; ++mt) {
      const int r = wm * 64 + mt * 16 + frow;
      const v8h a0 = *(const v8h*)&sA[buf][r * LDS_STRIDE + kselA];
      const v8h a1 = *(const v8h*)&sA[buf][r * LDS_STRIDE + kselA + 16];
      afrag[mt] = __builtin_shufflevector(a0, a1,
          0, 1, 2, 3, 4, 5, 6, 7, 8, 9, 10, 11, 12, 13, 14, 15);
    }
#pragma unroll
    for (int nt = 0; nt < 4; ++nt) {
      const int c = wn * 64 + nt * 16 + frow;
      const v8h b0 = *(const v8h*)&sB[buf][c * LDS_STRIDE + kselB];
      const v8h b1 = *(const v8h*)&sB[buf][c * LDS_STRIDE + kselB + 16];
      bfrag[nt] = __builtin_shufflevector(b0, b1,
          0, 1, 2, 3, 4, 5, 6, 7, 8, 9, 10, 11, 12, 13, 14, 15);
    }

#pragma unroll
    for (int mt = 0; mt < 4; ++mt)
#pragma unroll
      for (int nt = 0; nt < 4; ++nt)
        acc[mt][nt] = __builtin_amdgcn_wmma_f32_16x16x32_f16(
            false, afrag[mt], false, bfrag[nt],
            (short)0, acc[mt][nt], false, false);

    __syncthreads();                          // all reads of `buf` done
    buf ^= 1;
  }

  // Epilogue: D 16x16 -> lane col = lane&15, rows (lane>>4)*8 + v
  const int erow = (lane >> 4) * 8;
  const int ecol = lane & 15;
#pragma unroll
  for (int nt = 0; nt < 4; ++nt) {
    const size_t col = colBase + wn * 64 + nt * 16 + ecol;
    const float bv = bias[col];
#pragma unroll
    for (int mt = 0; mt < 4; ++mt)
#pragma unroll
      for (int v = 0; v < 8; ++v) {
        const size_t r = rowBase + wm * 64 + mt * 16 + erow + v;
        C[r * N_FEAT + col] = acc[mt][nt][v] + bv;
      }
  }
}

// ---------------------------------------------------------------------------
extern "C" void kernel_launch(void* const* d_in, const int* in_sizes, int n_in,
                              void* d_out, int out_size, void* d_ws, size_t ws_size,
                              hipStream_t stream)
{
  const float* x    = (const float*)d_in[0];   // [8192, 4096]
  const float* w    = (const float*)d_in[1];   // [4096, 4096]
  const float* bias = (const float*)d_in[2];   // [4096]
  float* out = (float*)d_out;                  // [8192, 4096]

  _Float16* w16 = (_Float16*)d_ws;                                   // 32 MB
  _Float16* x16 = (_Float16*)((char*)d_ws +
                    (size_t)N_FEAT * N_FEAT * sizeof(_Float16));     // 64 MB

  cvt_f32_f16<<<(unsigned)((size_t)N_FEAT * N_FEAT / (256 * 8)), 256, 0, stream>>>(w, w16);
  fwht_rows_f32_to_f16<<<TOKENS, 256, 0, stream>>>(x, x16);
  dim3 grid(N_FEAT / BN, TOKENS / BM);
  gemm_wmma_f16<<<grid, 256, 0, stream>>>(x16, w16, bias, out);
  fwht_rows_f32_inplace<<<TOKENS, 256, 0, stream>>>(out);
}
